// TransformerDecoderLayer_51427938403028
// MI455X (gfx1250) — matrix operand order repaired and checked by reading, not compiled, not run on hardware
//
#include <hip/hip_runtime.h>
#include <hip/hip_bf16.h>
#include <math.h>

// ---------------------------------------------------------------------------
// Types for CDNA5 WMMA (wave32): v_wmma_f32_16x16x32_f16
// ---------------------------------------------------------------------------
typedef _Float16 v16h __attribute__((ext_vector_type(16)));
typedef _Float16 v8h  __attribute__((ext_vector_type(8)));
typedef _Float16 v4h  __attribute__((ext_vector_type(4)));
typedef float    v8f  __attribute__((ext_vector_type(8)));
typedef float    v4f  __attribute__((ext_vector_type(4)));

union HFrag { v16h v; v8h h[2]; };

__device__ __forceinline__ v8f wmma16(const v16h& a, const v16h& b, const v8f& c) {
  // D = A(16x32 f16) * B(32x16 f16) + C(16x16 f32)
  return __builtin_amdgcn_wmma_f32_16x16x32_f16(false, a, false, b, (short)0, c,
                                                false, false);
}

// ---------------------------------------------------------------------------
// Problem constants (from reference): B=2, L=S=2048, D=1024, H=16, E=64, F=4096
// ---------------------------------------------------------------------------
#define LSEQ  2048
#define DMOD  1024
#define NHEAD 16
#define EHEAD 64
#define NTOK  4096        // B * L
#define FFDIM 4096
#define NBH   32          // B * H
#define UTOP  40          // FACTOR * ceil(ln(2048)) = 5*8

// ===========================================================================
// GEMM: C[M,N] = act(A[M,K] @ W[K,N] + bias),  M,N,K multiples of 128/128/32.
// Block 256 threads (8 wave32 waves), tile 128x128, BK=32, double-buffered LDS.
// Wave w: rows (w&3)*32..+32, cols (w>>2)*64..+64  -> 2x4 WMMA tiles.
// ===========================================================================
template <bool RELU>
__global__ void __launch_bounds__(256)
gemm_bias(const float* __restrict__ A, const float* __restrict__ W,
          const float* __restrict__ bias, float* __restrict__ C,
          int M, int N, int K)
{
  __shared__ _Float16 As[2][128][40];   // [m][k], pad 40 to dodge bank conflicts
  __shared__ _Float16 Bs[2][128][40];   // transposed: [n][k]

  const int tid  = threadIdx.x;
  const int w    = tid >> 5;
  const int lane = tid & 31;
  const int wm   = (w & 3) * 32;
  const int wn   = (w >> 2) * 64;
  const int bm   = blockIdx.y * 128;
  const int bn   = blockIdx.x * 128;

  v8f acc[2][4];
#pragma unroll
  for (int i = 0; i < 2; ++i)
#pragma unroll
    for (int j = 0; j < 4; ++j) { v8f z = {}; acc[i][j] = z; }

  int a_r[4], a_k[4], b_k[4], b_n[4];
#pragma unroll
  for (int i = 0; i < 4; ++i) {
    int v = tid + i * 256;
    a_r[i] = v >> 3;  a_k[i] = (v & 7) * 4;    // A tile: 128 rows x 8 float4
    b_k[i] = v >> 5;  b_n[i] = (v & 31) * 4;   // W tile: 32 rows x 32 float4
  }

  const int NK = K >> 5;
  v4f ra[4], rb[4];

  auto loadTile = [&](int kt) {
#pragma unroll
    for (int i = 0; i < 4; ++i) {
      ra[i] = *(const v4f*)(A + (size_t)(bm + a_r[i]) * K + kt * 32 + a_k[i]);
      rb[i] = *(const v4f*)(W + (size_t)(kt * 32 + b_k[i]) * N + bn + b_n[i]);
    }
  };
  auto storeTile = [&](int buf) {
#pragma unroll
    for (int i = 0; i < 4; ++i) {
      v4h ha = __builtin_convertvector(ra[i], v4h);
      *(v4h*)&As[buf][a_r[i]][a_k[i]] = ha;
      v4h hb = __builtin_convertvector(rb[i], v4h);
      Bs[buf][b_n[i] + 0][b_k[i]] = hb.x;
      Bs[buf][b_n[i] + 1][b_k[i]] = hb.y;
      Bs[buf][b_n[i] + 2][b_k[i]] = hb.z;
      Bs[buf][b_n[i] + 3][b_k[i]] = hb.w;
    }
  };

  loadTile(0);
  storeTile(0);
  __syncthreads();

  const int cbA = (lane >> 4) * 8;    // A-frag: K chunks {cbA..cbA+7, cbA+16..cbA+23}
  const int kbB = (lane >> 4) * 8;    // B-frag: K chunks {kbB..kbB+7, kbB+16..kbB+23}

  for (int kt = 0; kt < NK; ++kt) {
    const int buf = kt & 1;
    if (kt + 1 < NK) loadTile(kt + 1);

    HFrag af[2], bf[4];
#pragma unroll
    for (int tm = 0; tm < 2; ++tm) {
      int r = wm + tm * 16 + (lane & 15);
      af[tm].h[0] = *(const v8h*)&As[buf][r][cbA];
      af[tm].h[1] = *(const v8h*)&As[buf][r][cbA + 16];
    }
#pragma unroll
    for (int tn = 0; tn < 4; ++tn) {
      int c = wn + tn * 16 + (lane & 15);
      bf[tn].h[0] = *(const v8h*)&Bs[buf][c][kbB];
      bf[tn].h[1] = *(const v8h*)&Bs[buf][c][kbB + 16];
    }
#pragma unroll
    for (int tm = 0; tm < 2; ++tm)
#pragma unroll
      for (int tn = 0; tn < 4; ++tn)
        acc[tm][tn] = wmma16(af[tm].v, bf[tn].v, acc[tm][tn]);

    if (kt + 1 < NK) storeTile(1 - buf);
    __syncthreads();
  }

  // Epilogue: C layout -> VGPR i holds row i + (lane>>4)*8, col = lane&15
#pragma unroll
  for (int tm = 0; tm < 2; ++tm)
#pragma unroll
    for (int tn = 0; tn < 4; ++tn) {
      int col = bn + wn + tn * 16 + (lane & 15);
      float bv = bias ? bias[col] : 0.0f;
#pragma unroll
      for (int i = 0; i < 8; ++i) {
        int row = bm + wm + tm * 16 + i + ((lane >> 4) << 3);
        float v = acc[tm][tn][i] + bv;
        if (RELU) v = fmaxf(v, 0.0f);
        C[(size_t)row * N + col] = v;
      }
    }
}

// ===========================================================================
// Fused cross-attention (flash): out[b,l,h,:] = softmax(QK^T/8) V, no mask.
// Block = 128 threads (4 waves), 64-query tile, S-tiles of 64 keys.
// Q/K/V in (B*L, 1024) layout with head slice at col h*64.
// ===========================================================================
#define FD 72
__global__ void __launch_bounds__(128)
cross_attn_flash(const float* __restrict__ Q, const float* __restrict__ Kp,
                 const float* __restrict__ Vp, float* __restrict__ Out)
{
  __shared__ _Float16 Qs[64][FD];        // [q][e], pre-scaled by 1/8
  __shared__ _Float16 Ks[64][FD];        // [key][e]
  __shared__ _Float16 Vt[64][FD];        // [e][key]  (transposed for B-frags)
  __shared__ _Float16 Ps[4][16][FD];     // per-wave P bounce (C-layout -> A-layout)

  const int tid  = threadIdx.x;
  const int w    = tid >> 5;
  const int lane = tid & 31;
  const int bh   = blockIdx.y;
  const int b    = bh >> 4;
  const int h    = bh & 15;
  const int q0   = blockIdx.x * 64;
  const size_t headOff = (size_t)h * EHEAD;

  // ---- stage Q tile (scaled) ----
#pragma unroll
  for (int i = 0; i < 8; ++i) {
    int v = tid + i * 128;
    int r = v >> 4, e = (v & 15) * 4;
    v4f q = *(const v4f*)(Q + ((size_t)(b * LSEQ + q0 + r)) * DMOD + headOff + e);
    v4h hq;
    hq.x = (_Float16)(q.x * 0.125f); hq.y = (_Float16)(q.y * 0.125f);
    hq.z = (_Float16)(q.z * 0.125f); hq.w = (_Float16)(q.w * 0.125f);
    *(v4h*)&Qs[r][e] = hq;
  }
  __syncthreads();

  // persistent Q A-fragments (two K-steps over E=64)
  HFrag qa[2];
  {
    int r  = w * 16 + (lane & 15);
    int cb = (lane >> 4) * 8;
    qa[0].h[0] = *(const v8h*)&Qs[r][cb];
    qa[0].h[1] = *(const v8h*)&Qs[r][cb + 16];
    qa[1].h[0] = *(const v8h*)&Qs[r][32 + cb];
    qa[1].h[1] = *(const v8h*)&Qs[r][32 + cb + 16];
  }

  v8f o[4];
#pragma unroll
  for (int nt = 0; nt < 4; ++nt) { v8f z = {}; o[nt] = z; }
  float mrow[8], lrow[8];
#pragma unroll
  for (int i = 0; i < 8; ++i) { mrow[i] = -3.0e38f; lrow[i] = 0.0f; }

  for (int s0 = 0; s0 < LSEQ; s0 += 64) {
    // ---- stage K and V tiles ----
#pragma unroll
    for (int i = 0; i < 8; ++i) {
      int v = tid + i * 128;
      int r = v >> 4, e = (v & 15) * 4;
      size_t g = ((size_t)(b * LSEQ + s0 + r)) * DMOD + headOff + e;
      v4f kv = *(const v4f*)(Kp + g);
      v4h hk = __builtin_convertvector(kv, v4h);
      *(v4h*)&Ks[r][e] = hk;
      v4f vv = *(const v4f*)(Vp + g);
      Vt[e + 0][r] = (_Float16)vv.x;
      Vt[e + 1][r] = (_Float16)vv.y;
      Vt[e + 2][r] = (_Float16)vv.z;
      Vt[e + 3][r] = (_Float16)vv.w;
    }
    __syncthreads();

    // ---- S = (Q/8) K^T : 4 key-subtiles x 2 K-steps ----
    v8f sacc[4];
#pragma unroll
    for (int nt = 0; nt < 4; ++nt) {
      HFrag kb0, kb1;
      int key = nt * 16 + (lane & 15);
      int eb  = (lane >> 4) * 8;
      kb0.h[0] = *(const v8h*)&Ks[key][eb];
      kb0.h[1] = *(const v8h*)&Ks[key][eb + 16];
      kb1.h[0] = *(const v8h*)&Ks[key][32 + eb];
      kb1.h[1] = *(const v8h*)&Ks[key][32 + eb + 16];
      v8f s = {};
      s = wmma16(qa[0].v, kb0.v, s);
      s = wmma16(qa[1].v, kb1.v, s);
      sacc[nt] = s;
    }

    // ---- online softmax update (row stats live in 16-lane halves) ----
    float alpha[8];
#pragma unroll
    for (int i = 0; i < 8; ++i) {
      float mx = fmaxf(fmaxf(sacc[0][i], sacc[1][i]), fmaxf(sacc[2][i], sacc[3][i]));
      mx = fmaxf(mx, __shfl_xor(mx, 1, 32));
      mx = fmaxf(mx, __shfl_xor(mx, 2, 32));
      mx = fmaxf(mx, __shfl_xor(mx, 4, 32));
      mx = fmaxf(mx, __shfl_xor(mx, 8, 32));
      float mnew = fmaxf(mrow[i], mx);
      alpha[i] = __expf(mrow[i] - mnew);
      mrow[i] = mnew;
    }
    float ls[8];
#pragma unroll
    for (int i = 0; i < 8; ++i) ls[i] = 0.0f;
#pragma unroll
    for (int nt = 0; nt < 4; ++nt)
#pragma unroll
      for (int i = 0; i < 8; ++i) {
        float p = __expf(sacc[nt][i] - mrow[i]);
        ls[i] += p;
        int r = i + ((lane >> 4) << 3);
        Ps[w][r][nt * 16 + (lane & 15)] = (_Float16)p;   // in-order per wave
      }
#pragma unroll
    for (int i = 0; i < 8; ++i) {
      float t = ls[i];
      t += __shfl_xor(t, 1, 32);
      t += __shfl_xor(t, 2, 32);
      t += __shfl_xor(t, 4, 32);
      t += __shfl_xor(t, 8, 32);
      lrow[i] = lrow[i] * alpha[i] + t;
    }
#pragma unroll
    for (int nt = 0; nt < 4; ++nt)
#pragma unroll
      for (int i = 0; i < 8; ++i) o[nt][i] = o[nt][i] * alpha[i];

    // ---- P A-fragments (LDS bounce, no barrier: same-wave LDS is in-order) ----
    HFrag pa[2];
    {
      int r  = lane & 15;
      int cb = (lane >> 4) * 8;
      pa[0].h[0] = *(const v8h*)&Ps[w][r][cb];
      pa[0].h[1] = *(const v8h*)&Ps[w][r][cb + 16];
      pa[1].h[0] = *(const v8h*)&Ps[w][r][32 + cb];
      pa[1].h[1] = *(const v8h*)&Ps[w][r][32 + cb + 16];
    }
    // ---- O += P V ----
#pragma unroll
    for (int nt = 0; nt < 4; ++nt) {
      HFrag vb0, vb1;
      int e  = nt * 16 + (lane & 15);
      int sb = (lane >> 4) * 8;
      vb0.h[0] = *(const v8h*)&Vt[e][sb];
      vb0.h[1] = *(const v8h*)&Vt[e][sb + 16];
      vb1.h[0] = *(const v8h*)&Vt[e][32 + sb];
      vb1.h[1] = *(const v8h*)&Vt[e][32 + sb + 16];
      o[nt] = wmma16(pa[0].v, vb0.v, o[nt]);
      o[nt] = wmma16(pa[1].v, vb1.v, o[nt]);
    }
    __syncthreads();   // all reads of Ks/Vt done before next stage
  }

  // ---- normalize and store (B,L,H,E) flat ----
#pragma unroll
  for (int nt = 0; nt < 4; ++nt) {
    int e = nt * 16 + (lane & 15);
#pragma unroll
    for (int i = 0; i < 8; ++i) {
      int r = w * 16 + i + ((lane >> 4) << 3);
      Out[((size_t)(b * LSEQ + q0 + r)) * DMOD + headOff + e] = o[nt][i] / lrow[i];
    }
  }
}

// ===========================================================================
// jax.random.randint(key(42), (2048,40), 0, 2048) via threefry2x32.
// key = (0,42); span=2048 is 2^11 -> idx = lower_bits & 2047, lower_bits =
// second half of the (2*2048*40)-word draw: pair j uses counts (j, 81920+j),
// idx[j] = y1 & 2047.
// ===========================================================================
__device__ __forceinline__ unsigned rotl32(unsigned x, int r) {
  return (x << r) | (x >> (32 - r));
}
__device__ __forceinline__ void tf4(unsigned& x0, unsigned& x1,
                                    int a, int b, int c, int d) {
  x0 += x1; x1 = rotl32(x1, a); x1 ^= x0;
  x0 += x1; x1 = rotl32(x1, b); x1 ^= x0;
  x0 += x1; x1 = rotl32(x1, c); x1 ^= x0;
  x0 += x1; x1 = rotl32(x1, d); x1 ^= x0;
}
__global__ void __launch_bounds__(256)
idx_kernel(int* __restrict__ idx)
{
  int j = blockIdx.x * 256 + threadIdx.x;
  if (j >= LSEQ * UTOP) return;
  const unsigned k0 = 0u, k1 = 42u, k2 = k0 ^ k1 ^ 0x1BD11BDAu;
  unsigned x0 = (unsigned)j + k0;
  unsigned x1 = (unsigned)(LSEQ * UTOP + j) + k1;
  tf4(x0, x1, 13, 15, 26, 6);  x0 += k1; x1 += k2 + 1u;
  tf4(x0, x1, 17, 29, 16, 24); x0 += k2; x1 += k0 + 2u;
  tf4(x0, x1, 13, 15, 26, 6);  x0 += k0; x1 += k1 + 3u;
  tf4(x0, x1, 17, 29, 16, 24); x0 += k1; x1 += k2 + 4u;
  tf4(x0, x1, 13, 15, 26, 6);  x0 += k2; x1 += k0 + 5u;
  idx[j] = (int)(x1 & 2047u);
}

// ===========================================================================
// M[b,h,l] = max_u q.k[idx[l,u]] - (sum_u q.k[idx[l,u]]) / 2048   (no 1/sqrt(E))
// ===========================================================================
__global__ void __launch_bounds__(256)
mscore_kernel(const float* __restrict__ q, const float* __restrict__ k,
              const int* __restrict__ idx, float* __restrict__ Msc)
{
  int t  = blockIdx.x * 256 + threadIdx.x;  // < 65536
  int bh = t >> 11, l = t & 2047;
  int b  = bh >> 4, h = bh & 15;
  const float* qr = q + ((size_t)(b * LSEQ + l)) * DMOD + h * EHEAD;
  v4f qv[16];
#pragma unroll
  for (int i = 0; i < 16; ++i) qv[i] = *(const v4f*)(qr + i * 4);
  float mx = -3.0e38f, sm = 0.0f;
  for (int u = 0; u < UTOP; ++u) {
    int s = idx[l * UTOP + u];
    const float* kr = k + ((size_t)(b * LSEQ + s)) * DMOD + h * EHEAD;
    float d = 0.0f;
#pragma unroll
    for (int i = 0; i < 16; ++i) {
      v4f kv = *(const v4f*)(kr + i * 4);
      d += qv[i].x * kv.x + qv[i].y * kv.y + qv[i].z * kv.z + qv[i].w * kv.w;
    }
    mx = fmaxf(mx, d);
    sm += d;
  }
  Msc[(size_t)bh * LSEQ + l] = mx - sm * (1.0f / (float)LSEQ);
}

// ===========================================================================
// Top-40 per (b,h): iterative argmax, JAX tie rule (equal -> smaller index).
// ===========================================================================
__global__ void __launch_bounds__(256)
topk_kernel(const float* __restrict__ Msc, int* __restrict__ Mtop)
{
  __shared__ float vals[LSEQ];
  __shared__ float rv[256];
  __shared__ int   ri[256];
  int bh = blockIdx.x, tid = threadIdx.x;
  for (int l = tid; l < LSEQ; l += 256) vals[l] = Msc[(size_t)bh * LSEQ + l];
  __syncthreads();
  for (int it = 0; it < UTOP; ++it) {
    float bv = -3.0e38f;
    int   bi = 1 << 30;
    for (int l = tid; l < LSEQ; l += 256) {
      float v = vals[l];
      if (v > bv || (v == bv && l < bi)) { bv = v; bi = l; }
    }
    rv[tid] = bv; ri[tid] = bi;
    __syncthreads();
    for (int st = 128; st > 0; st >>= 1) {
      if (tid < st) {
        float ov = rv[tid + st]; int oi = ri[tid + st];
        if (ov > rv[tid] || (ov == rv[tid] && oi < ri[tid])) {
          rv[tid] = ov; ri[tid] = oi;
        }
      }
      __syncthreads();
    }
    if (tid == 0) { Mtop[bh * UTOP + it] = ri[0]; vals[ri[0]] = -3.0e38f; }
    __syncthreads();
  }
}

// ---- block reduction helpers (256 threads) ----
__device__ __forceinline__ float blockSum(float v, float* red) {
  red[threadIdx.x] = v; __syncthreads();
  for (int st = 128; st > 0; st >>= 1) {
    if (threadIdx.x < st) red[threadIdx.x] += red[threadIdx.x + st];
    __syncthreads();
  }
  float r = red[0]; __syncthreads();
  return r;
}
__device__ __forceinline__ float blockMax(float v, float* red) {
  red[threadIdx.x] = v; __syncthreads();
  for (int st = 128; st > 0; st >>= 1) {
    if (threadIdx.x < st) red[threadIdx.x] = fmaxf(red[threadIdx.x], red[threadIdx.x + st]);
    __syncthreads();
  }
  float r = red[0]; __syncthreads();
  return r;
}

// ===========================================================================
// Per selected query u: scores = Q_red.k / 8 with mask (s > m -> -1e9),
// softmax, upd = attn @ v.   One block per (u, bh).
// ===========================================================================
__global__ void __launch_bounds__(256)
prob_attn_kernel(const float* __restrict__ q, const float* __restrict__ k,
                 const float* __restrict__ v, const int* __restrict__ Mtop,
                 float* __restrict__ upd)
{
  __shared__ float sc[LSEQ];
  __shared__ float qsh[EHEAD];
  __shared__ float red[256];
  __shared__ float pacc[4][EHEAD];
  int u = blockIdx.x, bh = blockIdx.y;
  int b = bh >> 4, h = bh & 15;
  int m = Mtop[bh * UTOP + u];
  int tid = threadIdx.x;
  if (tid < EHEAD)
    qsh[tid] = q[((size_t)(b * LSEQ + m)) * DMOD + h * EHEAD + tid];
  __syncthreads();

  float lm = -3.0e38f;
  for (int s = tid; s < LSEQ; s += 256) {
    float d;
    if (s > m) d = -1.0e9f;
    else {
      d = 0.0f;
      const float* kr = k + ((size_t)(b * LSEQ + s)) * DMOD + h * EHEAD;
#pragma unroll 8
      for (int e = 0; e < EHEAD; ++e) d += qsh[e] * kr[e];
      d *= 0.125f;
    }
    sc[s] = d;
    lm = fmaxf(lm, d);
  }
  __syncthreads();
  float M = blockMax(lm, red);

  float lsum = 0.0f;
  for (int s = tid; s < LSEQ; s += 256) {
    float p = __expf(sc[s] - M);
    sc[s] = p;
    lsum += p;
  }
  __syncthreads();
  float Ssum = blockSum(lsum, red);

  int e = tid & 63, c = tid >> 6;
  float acc = 0.0f;
  for (int s = c * 512; s < (c + 1) * 512; ++s)
    acc += sc[s] * v[((size_t)(b * LSEQ + s)) * DMOD + h * EHEAD + e];
  pacc[c][e] = acc;
  __syncthreads();
  if (tid < EHEAD) {
    float r = (pacc[0][tid] + pacc[1][tid] + pacc[2][tid] + pacc[3][tid]) / Ssum;
    upd[((size_t)bh * UTOP + u) * EHEAD + tid] = r;
  }
}

// ctx[b,l,h,e] = inclusive cumsum over l of v[b,l,h,e]   ((B*L, D) layout)
__global__ void __launch_bounds__(64)
cumsum_kernel(const float* __restrict__ v, float* __restrict__ ctx)
{
  int bh = blockIdx.x, e = threadIdx.x;
  int b = bh >> 4, h = bh & 15;
  size_t off = (size_t)b * LSEQ * DMOD + h * EHEAD + e;
  float acc = 0.0f;
  for (int l = 0; l < LSEQ; ++l) {
    acc += v[off + (size_t)l * DMOD];
    ctx[off + (size_t)l * DMOD] = acc;
  }
}

// ctx[b, Mtop[u], h, :] = upd[bh, u, :]
__global__ void __launch_bounds__(256)
scatter_kernel(const float* __restrict__ upd, const int* __restrict__ Mtop,
               float* __restrict__ ctx)
{
  int bh = blockIdx.x, tid = threadIdx.x;
  int b = bh >> 4, h = bh & 15;
  for (int i = tid; i < UTOP * EHEAD; i += 256) {
    int u = i >> 6, e = i & 63;
    int m = Mtop[bh * UTOP + u];
    ctx[((size_t)(b * LSEQ + m)) * DMOD + h * EHEAD + e] =
        upd[((size_t)bh * UTOP + u) * EHEAD + e];
  }
}

// out = LayerNorm(a + b) * g + be   (row length 1024, 256 threads, 4 elem/thr)
__global__ void __launch_bounds__(256)
add_ln_kernel(const float* __restrict__ a, const float* __restrict__ b,
              const float* __restrict__ g, const float* __restrict__ be,
              float* __restrict__ out)
{
  __shared__ float red[256];
  const int row = blockIdx.x, tid = threadIdx.x;
  const int c = tid * 4;
  const size_t base = (size_t)row * DMOD + c;
  v4f va = *(const v4f*)(a + base);
  v4f vb = *(const v4f*)(b + base);
  float x0 = va.x + vb.x, x1 = va.y + vb.y, x2 = va.z + vb.z, x3 = va.w + vb.w;
  float s = blockSum(x0 + x1 + x2 + x3, red);
  float mean = s * (1.0f / (float)DMOD);
  float d0 = x0 - mean, d1 = x1 - mean, d2 = x2 - mean, d3 = x3 - mean;
  float sq = blockSum(d0 * d0 + d1 * d1 + d2 * d2 + d3 * d3, red);
  float rstd = rsqrtf(sq * (1.0f / (float)DMOD) + 1e-5f);
  v4f vg = *(const v4f*)(g + c);
  v4f vbe = *(const v4f*)(be + c);
  v4f r;
  r.x = d0 * rstd * vg.x + vbe.x;
  r.y = d1 * rstd * vg.y + vbe.y;
  r.z = d2 * rstd * vg.z + vbe.z;
  r.w = d3 * rstd * vg.w + vbe.w;
  *(v4f*)(out + base) = r;
}

// ===========================================================================
// Orchestration
// ===========================================================================
extern "C" void kernel_launch(void* const* d_in, const int* in_sizes, int n_in,
                              void* d_out, int out_size, void* d_ws, size_t ws_size,
                              hipStream_t stream)
{
  const float* x      = (const float*)d_in[0];
  const float* memory = (const float*)d_in[1];
  const float* sWq = (const float*)d_in[2];  const float* sbq = (const float*)d_in[3];
  const float* sWk = (const float*)d_in[4];  const float* sbk = (const float*)d_in[5];
  const float* sWv = (const float*)d_in[6];  const float* sbv = (const float*)d_in[7];
  const float* sWo = (const float*)d_in[8];  const float* sbo = (const float*)d_in[9];
  const float* cWq = (const float*)d_in[10]; const float* cbq = (const float*)d_in[11];
  const float* cWk = (const float*)d_in[12]; const float* cbk = (const float*)d_in[13];
  const float* cWv = (const float*)d_in[14]; const float* cbv = (const float*)d_in[15];
  const float* cWo = (const float*)d_in[16]; const float* cbo = (const float*)d_in[17];
  const float* W1  = (const float*)d_in[18]; const float* b1  = (const float*)d_in[19];
  const float* W2  = (const float*)d_in[20]; const float* b2  = (const float*)d_in[21];
  const float* g1  = (const float*)d_in[22]; const float* be1 = (const float*)d_in[23];
  const float* g2  = (const float*)d_in[24]; const float* be2 = (const float*)d_in[25];
  const float* g3  = (const float*)d_in[26]; const float* be3 = (const float*)d_in[27];

  float* ws = (float*)d_ws;
  const size_t P = (size_t)NTOK * DMOD;          // 4,194,304 floats
  float* qs   = ws;                               // reused as qc later
  float* ks   = ws + P;                           // reused as kc
  float* vs   = ws + 2 * P;                       // reused as vc
  float* ctx  = ws + 3 * P;                       // self ctx / cross attn out
  float* tmp  = ws + 4 * P;                       // proj outputs (sa / ca / y)
  float* x1   = ws + 5 * P;
  float* x2   = ws + 6 * P;
  float* ffn  = ws + 7 * P;                       // (4096, 4096)
  float* msc  = ffn + (size_t)NTOK * FFDIM;       // 65536
  float* updp = msc + (size_t)NBH * LSEQ;         // 32*40*64
  int*   idxp = (int*)(updp + (size_t)NBH * UTOP * EHEAD);
  int*   mtop = idxp + LSEQ * UTOP;

  const dim3 blk(256);
  const dim3 gDD(DMOD / 128, NTOK / 128);         // (8, 32)
  const dim3 gDF(FFDIM / 128, NTOK / 128);        // (32, 32)

  // ---- self-attention projections ----
  gemm_bias<false><<<gDD, blk, 0, stream>>>(x, sWq, sbq, qs, NTOK, DMOD, DMOD);
  gemm_bias<false><<<gDD, blk, 0, stream>>>(x, sWk, sbk, ks, NTOK, DMOD, DMOD);
  gemm_bias<false><<<gDD, blk, 0, stream>>>(x, sWv, sbv, vs, NTOK, DMOD, DMOD);

  // ---- ProbSparse self-attention ----
  idx_kernel<<<(LSEQ * UTOP + 255) / 256, blk, 0, stream>>>(idxp);
  mscore_kernel<<<(NBH * LSEQ) / 256, blk, 0, stream>>>(qs, ks, idxp, msc);
  topk_kernel<<<NBH, blk, 0, stream>>>(msc, mtop);
  prob_attn_kernel<<<dim3(UTOP, NBH), blk, 0, stream>>>(qs, ks, vs, mtop, updp);
  cumsum_kernel<<<NBH, 64, 0, stream>>>(vs, ctx);
  scatter_kernel<<<NBH, blk, 0, stream>>>(updp, mtop, ctx);
  gemm_bias<false><<<gDD, blk, 0, stream>>>(ctx, sWo, sbo, tmp, NTOK, DMOD, DMOD);
  add_ln_kernel<<<NTOK, blk, 0, stream>>>(x, tmp, g1, be1, x1);

  // ---- cross attention (fused flash, WMMA) ----
  gemm_bias<false><<<gDD, blk, 0, stream>>>(x1, cWq, cbq, qs, NTOK, DMOD, DMOD);
  gemm_bias<false><<<gDD, blk, 0, stream>>>(memory, cWk, cbk, ks, NTOK, DMOD, DMOD);
  gemm_bias<false><<<gDD, blk, 0, stream>>>(memory, cWv, cbv, vs, NTOK, DMOD, DMOD);
  cross_attn_flash<<<dim3(LSEQ / 64, NBH), 128, 0, stream>>>(qs, ks, vs, ctx);
  gemm_bias<false><<<gDD, blk, 0, stream>>>(ctx, cWo, cbo, tmp, NTOK, DMOD, DMOD);
  add_ln_kernel<<<NTOK, blk, 0, stream>>>(x1, tmp, g2, be2, x2);

  // ---- FFN ----
  gemm_bias<true ><<<gDF, blk, 0, stream>>>(x2, W1, b1, ffn, NTOK, FFDIM, DMOD);
  gemm_bias<false><<<gDD, blk, 0, stream>>>(ffn, W2, b2, tmp, NTOK, DMOD, FFDIM);
  add_ln_kernel<<<NTOK, blk, 0, stream>>>(x2, tmp, g3, be3, (float*)d_out);
}
